// LinearSelfAttention_32693291057187
// MI455X (gfx1250) — compile-verified, matrix operand order checked by reading, not run on hardware
//
#include <hip/hip_runtime.h>
#include <hip/hip_bf16.h>

// ---------------------------------------------------------------------------
// LinearSelfAttention fused pipeline for gfx1250 (MI455X, wave32, WMMA + TDM).
//   B=4, N=4096, M=4096, DIM=1024, HEADS=8, DH=128, NM=8192
// GEMMs run on v_wmma_f32_16x16x32_bf16; the A tile is DMA'd into LDS by the
// Tensor Data Mover (tensor_load_to_lds, TENSORcnt-fenced); B is staged
// transposed so both fragments load as 2x ds_load_b128 per the CDNA5 16-bit
// WMMA VGPR layout (lane-half h holds K 8h..8h+7 and 16+8h..+7).
// ---------------------------------------------------------------------------

typedef __bf16 bf16;
typedef __attribute__((ext_vector_type(16))) __bf16 v16bf;
typedef __attribute__((ext_vector_type(8)))  float  v8f;
typedef __attribute__((ext_vector_type(4)))  unsigned uint32x4;
typedef __attribute__((ext_vector_type(4)))  int      int32x4;
typedef __attribute__((ext_vector_type(8)))  int      int32x8;

union FragBF { v16bf v; uint4 q[2]; };

#define TM 128
#define TN 128
#define TK 32
#define T_LD 48   // 48 elems * 2B = 96B row stride (16B aligned, TDM-pad compatible)

// ---- WMMA 16-bit fragment gather (A and B share the K enumeration) ---------
// lane L: row/col = base + L%16, half = L/16.
// VGPRs 0..3 hold K = half*8 .. +7 ; VGPRs 4..7 hold K = 16+half*8 .. +7
// -> two contiguous 16-byte chunks when K is the fastest LDS dimension.
__device__ __forceinline__ v16bf load_frag(const bf16 (*T)[T_LD], int base, int lane) {
    FragBF f;
    const int r  = base + (lane & 15);
    const int h8 = (lane >> 4) << 3;
    f.q[0] = *reinterpret_cast<const uint4*>(&T[r][h8]);
    f.q[1] = *reinterpret_cast<const uint4*>(&T[r][16 + h8]);
    return f.v;
}

// ---- Tensor Data Mover: 2D tile (tileK x tileM) of bf16 into LDS -----------
// Row = tileK elems; pad_interval=16 DWORDs (64B row), pad_amount=8 DWORDs
// (32B) -> LDS row stride 48 elements == T_LD.
__device__ __forceinline__ void tdm_load_tile_a(
    const bf16* gptr, long long lda_elems, int tensorK, int tensorM,
    unsigned lds_byte_off)
{
    const unsigned long long ga = (unsigned long long)gptr;
    uint32x4 g0;
    g0.x = 1u;                                   // count=1, user descriptor
    g0.y = lds_byte_off;                         // lds_addr
    g0.z = (unsigned)ga;                         // global_addr[31:0]
    g0.w = (unsigned)((ga >> 32) & 0x01FFFFFFu) | (2u << 30);  // addr[56:32] | type=2

    const unsigned kdim = (unsigned)tensorK;
    const unsigned mdim = (unsigned)tensorM;
    const unsigned long long stride = (unsigned long long)lda_elems;
    int32x8 g1;
    g1[0] = (int)((1u << 16)      // data_size = 2 bytes
                | (1u << 20)      // pad_enable
                | (3u << 22)      // pad_interval: 16 DWORDs
                | (7u << 25));    // pad_amount : 8 DWORDs
    g1[1] = (int)((kdim & 0xFFFFu) << 16);                       // tensor_dim0 lo
    g1[2] = (int)((kdim >> 16) | ((mdim & 0xFFFFu) << 16));      // dim0 hi | dim1 lo
    g1[3] = (int)((mdim >> 16) | ((unsigned)TK << 16));          // dim1 hi | tile_dim0=32
    g1[4] = (int)TM;                                             // tile_dim1=128, tile_dim2=0
    g1[5] = (int)(unsigned)(stride & 0xFFFFFFFFull);             // tensor_dim0_stride lo
    g1[6] = (int)(unsigned)((stride >> 32) & 0xFFFFu);           // stride hi | dim1_stride lo
    g1[7] = 0;

    const int32x4 gz = {0, 0, 0, 0};
#if __clang_major__ >= 23
    const int32x8 gz8 = {0, 0, 0, 0, 0, 0, 0, 0};
    __builtin_amdgcn_tensor_load_to_lds(g0, g1, gz, gz, gz8, 0);
#else
    __builtin_amdgcn_tensor_load_to_lds(g0, g1, gz, gz, 0);
#endif
}

// ---- Generic batched bf16 GEMM: C[M,N] = A[M,K] * B[K,N], f32 accum --------
// grid = (N/128, M/128, batch); block = 256 threads (8 wave32 waves).
__global__ __launch_bounds__(256) void gemm_bf16_wmma(
    const bf16* __restrict__ A, long long lda, long long strideA,
    const bf16* __restrict__ B, long long ldb, long long strideB,
    float*      __restrict__ C, long long ldc, long long strideC,
    int M, int N, int K)
{
    __shared__ bf16 As[TM][T_LD];   // [row][k]
    __shared__ bf16 Bt[TN][T_LD];   // [col][k]  (B staged transposed)

    A += (long long)blockIdx.z * strideA;
    B += (long long)blockIdx.z * strideB;
    C += (long long)blockIdx.z * strideC;

    const int m0   = blockIdx.y * TM;
    const int n0   = blockIdx.x * TN;
    const int tid  = threadIdx.x;
    const int lane = tid & 31;
    const int wave = tid >> 5;   // 0..7
    const int wm   = wave >> 1;  // 0..3 : 32-row strip
    const int wn   = wave & 1;   // 0..1 : 64-col strip

    const unsigned as_lds_off = (unsigned)(unsigned long long)(const void*)&As[0][0];

    v8f acc[2][4];
#pragma unroll
    for (int i = 0; i < 2; ++i)
#pragma unroll
        for (int j = 0; j < 4; ++j) acc[i][j] = v8f{};

    for (int k0 = 0; k0 < K; k0 += TK) {
        __syncthreads();   // previous iteration's fragment reads complete

        // A tile via Tensor Data Mover (one issue per block, wave 0)
        if (wave == 0) {
            tdm_load_tile_a(A + (long long)m0 * lda + k0, lda, K, M, as_lds_off);
        }

        // B tile staged transposed: Bt[col][k] = B[(k0+k)*ldb + n0+col]
        // Adjacent lanes read adjacent columns -> coalesced global_load_u16.
        {
            const int col = tid & 127;
            const int kh  = (tid >> 7) << 4;   // 0 or 16
            alignas(16) bf16 tmp[16];
#pragma unroll
            for (int kk = 0; kk < 16; ++kk)
                tmp[kk] = B[(long long)(k0 + kh + kk) * ldb + n0 + col];
            *reinterpret_cast<uint4*>(&Bt[col][kh])     = *reinterpret_cast<const uint4*>(&tmp[0]);
            *reinterpret_cast<uint4*>(&Bt[col][kh + 8]) = *reinterpret_cast<const uint4*>(&tmp[8]);
        }

        if (k0 + TK < K) {  // hint next B tile toward L2 (global_prefetch_b8)
            __builtin_prefetch(B + (long long)(k0 + TK + (tid & 31)) * ldb + n0, 0, 1);
        }

        if (wave == 0) {
            __builtin_amdgcn_s_wait_tensorcnt(0);   // TDM writes visible in LDS
        }
        __syncthreads();

        v16bf afrag[2], bfrag[4];
#pragma unroll
        for (int i = 0; i < 2; ++i) afrag[i] = load_frag(As, wm * 32 + i * 16, lane);
#pragma unroll
        for (int j = 0; j < 4; ++j) bfrag[j] = load_frag(Bt, wn * 64 + j * 16, lane);

#pragma unroll
        for (int i = 0; i < 2; ++i)
#pragma unroll
            for (int j = 0; j < 4; ++j)
                acc[i][j] = __builtin_amdgcn_wmma_f32_16x16x32_bf16(
                    /*neg_a=*/false, afrag[i], /*neg_b=*/false, bfrag[j],
                    /*c_mod=*/(short)0, acc[i][j], /*reuse_a=*/false, /*reuse_b=*/false);
    }

    // C/D 16x16 f32 layout: lane L -> N = L%16; VGPR r -> M = r + 8*(L/16)
#pragma unroll
    for (int i = 0; i < 2; ++i) {
        const int mbase = m0 + wm * 32 + i * 16 + ((lane >> 4) << 3);
#pragma unroll
        for (int j = 0; j < 4; ++j) {
            const int ncol = n0 + wn * 64 + j * 16 + (lane & 15);
#pragma unroll
            for (int r = 0; r < 8; ++r)
                C[(long long)(mbase + r) * ldc + ncol] = acc[i][j][r];
        }
    }
}

// ---- LayerNorm (1024-wide rows) -> bf16 into kv_input[b, n, :] -------------
__global__ __launch_bounds__(256) void layernorm_cast_kernel(
    const float* __restrict__ x, bf16* __restrict__ kvin)
{
    const int row = blockIdx.x;        // b*4096 + n
    const int b   = row >> 12;
    const int n   = row & 4095;
    const float* xr = x + (long long)row * 1024;

    __shared__ float red[2][8];
    float vals[4];
    float s = 0.f, s2 = 0.f;
#pragma unroll
    for (int i = 0; i < 4; ++i) {
        const float v = xr[threadIdx.x + i * 256];
        vals[i] = v; s += v; s2 += v * v;
    }
#pragma unroll
    for (int off = 16; off; off >>= 1) {
        s  += __shfl_xor(s,  off, 32);
        s2 += __shfl_xor(s2, off, 32);
    }
    const int wave = threadIdx.x >> 5, lane = threadIdx.x & 31;
    if (lane == 0) { red[0][wave] = s; red[1][wave] = s2; }
    __syncthreads();
    float ts = 0.f, ts2 = 0.f;
#pragma unroll
    for (int i = 0; i < 8; ++i) { ts += red[0][i]; ts2 += red[1][i]; }
    const float mu   = ts * (1.f / 1024.f);
    const float var  = ts2 * (1.f / 1024.f) - mu * mu;
    const float rsig = rsqrtf(var + 1e-5f);

    bf16* o = kvin + ((long long)b * 8192 + n) * 1024;
#pragma unroll
    for (int i = 0; i < 4; ++i)
        o[threadIdx.x + i * 256] = (bf16)((vals[i] - mu) * rsig);
}

// ---- hiddens (f32) -> bf16 into kv_input[b, 4096+n, :] ---------------------
__global__ __launch_bounds__(256) void cast_hiddens_kernel(
    const float* __restrict__ h, bf16* __restrict__ kvin)
{
    const long long i = (long long)blockIdx.x * 256 + threadIdx.x;  // 4*4096*1024
    const long long b = i >> 22;
    const long long r = i & ((1LL << 22) - 1);
    kvin[((b * 8192) + 4096) * 1024 + r] = (bf16)h[i];
}

// ---- generic f32 -> bf16 cast ----------------------------------------------
__global__ __launch_bounds__(256) void cast_f32_bf16_kernel(
    const float* __restrict__ in, bf16* __restrict__ out, long long n)
{
    long long i = (long long)blockIdx.x * 256 + threadIdx.x;
    const long long stride = (long long)gridDim.x * 256;
    for (; i < n; i += stride) out[i] = (bf16)in[i];
}

// ---- softmax over feature dim (per head, 128 wide); q -> [b,h,n,128] bf16 --
__global__ __launch_bounds__(256) void softmax_q_kernel(
    const float* __restrict__ q, bf16* __restrict__ qb)
{
    const int row  = blockIdx.x;          // b*4096 + n
    const int b    = row >> 12;
    const int n    = row & 4095;
    const int h    = threadIdx.x >> 5;    // wave == head
    const int lane = threadIdx.x & 31;
    const float* src = q + (long long)row * 1024 + h * 128;

    float v[4]; float mx = -3.4e38f;
#pragma unroll
    for (int i = 0; i < 4; ++i) { v[i] = src[lane + i * 32]; mx = fmaxf(mx, v[i]); }
#pragma unroll
    for (int off = 16; off; off >>= 1) mx = fmaxf(mx, __shfl_xor(mx, off, 32));
    float sum = 0.f;
#pragma unroll
    for (int i = 0; i < 4; ++i) { v[i] = __expf(v[i] - mx); sum += v[i]; }
#pragma unroll
    for (int off = 16; off; off >>= 1) sum += __shfl_xor(sum, off, 32);
    const float r = 1.f / sum;

    bf16* dst = qb + (((long long)(b * 8 + h) * 4096) + n) * 128;
#pragma unroll
    for (int i = 0; i < 4; ++i) dst[lane + i * 32] = (bf16)(v[i] * r);
}

// ---- softmax over sequence dim on k columns + transpose --------------------
// kv f32 [b, 8192, 2048]; k = cols 0..1023. Output kt bf16 [b,h,128,8192].
__global__ __launch_bounds__(256) void softmax_k_kernel(
    const float* __restrict__ kv, bf16* __restrict__ kt)
{
    const int idx = blockIdx.x * 256 + threadIdx.x;   // 0..4095 == (b, c)
    const int b = idx >> 10;
    const int c = idx & 1023;
    const int h = c >> 7;
    const int d = c & 127;
    const float* col = kv + (long long)b * 8192 * 2048 + c;

    float mx = -3.4e38f;
    for (int n = 0; n < 8192; ++n) mx = fmaxf(mx, col[(long long)n * 2048]);
    float sum = 0.f;
    for (int n = 0; n < 8192; ++n) sum += __expf(col[(long long)n * 2048] - mx);
    const float r = 1.f / sum;

    bf16* dst = kt + (((long long)(b * 8 + h) * 128) + d) * 8192;
    for (int n = 0; n < 8192; ++n)
        dst[n] = (bf16)(__expf(col[(long long)n * 2048] - mx) * r);
}

// ---- v rearrange: kv f32 [b,8192,2048] cols 1024.. -> v bf16 [b,h,8192,128]
__global__ __launch_bounds__(256) void rearrange_v_kernel(
    const float* __restrict__ kv, bf16* __restrict__ vb)
{
    const long long i = (long long)blockIdx.x * 256 + threadIdx.x; // 4*8192*1024
    const long long bnm = i >> 10;          // b*8192 + n
    const int c = (int)(i & 1023);
    const int h = c >> 7, e = c & 127;
    const long long b = bnm >> 13, n = bnm & 8191;
    vb[(((b * 8 + h) * 8192) + n) * 128 + e] = (bf16)kv[bnm * 2048 + 1024 + c];
}

// ---- merge heads: attn f32 [b,h,n,128] -> bf16 [b,n,1024] ------------------
__global__ __launch_bounds__(256) void merge_attn_kernel(
    const float* __restrict__ a, bf16* __restrict__ o)
{
    const long long i = (long long)blockIdx.x * 256 + threadIdx.x; // 4*8*4096*128
    const int e = (int)(i & 127);
    const int n = (int)((i >> 7) & 4095);
    const int h = (int)((i >> 19) & 7);
    const long long b = i >> 22;
    o[((b * 4096 + n) * 1024) + h * 128 + e] = (bf16)a[i];
}

// ---------------------------------------------------------------------------
extern "C" void kernel_launch(void* const* d_in, const int* in_sizes, int n_in,
                              void* d_out, int out_size, void* d_ws, size_t ws_size,
                              hipStream_t stream) {
    const float* x     = (const float*)d_in[0];  // [4,4096,1024]
    const float* hid   = (const float*)d_in[1];  // [4,4096,1024]
    const float* w_q   = (const float*)d_in[2];  // [1024,1024]
    const float* w_kv  = (const float*)d_in[3];  // [1024,2048]
    const float* w_out = (const float*)d_in[4];  // [1024,1024]
    float* out = (float*)d_out;                  // [4,4096,1024] f32

    (void)in_sizes; (void)n_in; (void)out_size; (void)ws_size;

    char* ws = (char*)d_ws;
    size_t off = 0;
    auto alloc = [&](size_t bytes) -> char* {
        char* p = ws + off;
        off += (bytes + 255) & ~(size_t)255;
        return p;
    };

    bf16* kvin   = (bf16*)alloc(4ULL * 8192 * 1024 * 2);        // [4,8192,1024] bf16
    bf16* wq_b   = (bf16*)alloc(1024ULL * 1024 * 2);
    bf16* wkv_b  = (bf16*)alloc(1024ULL * 2048 * 2);
    bf16* wout_b = (bf16*)alloc(1024ULL * 1024 * 2);
    float* q_f32 = (float*)alloc(4ULL * 4096 * 1024 * 4);       // [4,4096,1024]
    float* kv_f32= (float*)alloc(4ULL * 8192 * 2048 * 4);       // [4,8192,2048]
    bf16* q_b    = (bf16*)alloc(32ULL * 4096 * 128 * 2);        // [b,h,4096,128]
    bf16* kt_b   = (bf16*)alloc(32ULL * 128 * 8192 * 2);        // [b,h,128,8192]
    bf16* v_b    = (bf16*)alloc(32ULL * 8192 * 128 * 2);        // [b,h,8192,128]
    float* ctx_f = (float*)alloc(32ULL * 128 * 128 * 4);        // [b,h,128,128]
    bf16* ctx_b  = (bf16*)alloc(32ULL * 128 * 128 * 2);
    bf16* attn_b = (bf16*)alloc(4ULL * 4096 * 1024 * 2);        // [b,n,1024]
    float* attn_f= q_f32;  // q_f32 dead after softmax_q; reuse for attn [b,h,n,128]

    // 1-2: LayerNorm(x) and hiddens cast into kv_input
    layernorm_cast_kernel<<<dim3(16384), dim3(256), 0, stream>>>(x, kvin);
    cast_hiddens_kernel<<<dim3(65536), dim3(256), 0, stream>>>(hid, kvin);

    // weight casts
    cast_f32_bf16_kernel<<<dim3(4096), dim3(256), 0, stream>>>(w_q,   wq_b,   1024LL * 1024);
    cast_f32_bf16_kernel<<<dim3(8192), dim3(256), 0, stream>>>(w_kv,  wkv_b,  1024LL * 2048);
    cast_f32_bf16_kernel<<<dim3(4096), dim3(256), 0, stream>>>(w_out, wout_b, 1024LL * 1024);

    // 3: q = normed @ w_q   (per-batch: A rows live inside kvin at b*8192)
    gemm_bf16_wmma<<<dim3(8, 32, 4), dim3(256), 0, stream>>>(
        kvin, 1024, 8192LL * 1024,
        wq_b, 1024, 0,
        q_f32, 1024, 4096LL * 1024,
        4096, 1024, 1024);

    // 4: kv = kv_input @ w_kv  (flattened 32768 rows)
    gemm_bf16_wmma<<<dim3(16, 256, 1), dim3(256), 0, stream>>>(
        kvin, 1024, 0,
        wkv_b, 2048, 0,
        kv_f32, 2048, 0,
        32768, 2048, 1024);

    // 5: softmax(q) over dh per head
    softmax_q_kernel<<<dim3(16384), dim3(256), 0, stream>>>(q_f32, q_b);

    // 6: softmax(k) over sequence + transpose
    softmax_k_kernel<<<dim3(16), dim3(256), 0, stream>>>(kv_f32, kt_b);

    // 7: rearrange v
    rearrange_v_kernel<<<dim3(131072), dim3(256), 0, stream>>>(kv_f32, v_b);

    // 8: ctx = k^T @ v   (32 batches of 128x128x8192)
    gemm_bf16_wmma<<<dim3(1, 1, 32), dim3(256), 0, stream>>>(
        kt_b, 8192, 128LL * 8192,
        v_b, 128, 8192LL * 128,
        ctx_f, 128, 128LL * 128,
        128, 128, 8192);

    cast_f32_bf16_kernel<<<dim3(2048), dim3(256), 0, stream>>>(ctx_f, ctx_b, 32LL * 128 * 128);

    // 9: attn = q @ ctx  (32 batches of 4096x128x128)
    gemm_bf16_wmma<<<dim3(1, 32, 32), dim3(256), 0, stream>>>(
        q_b, 128, 4096LL * 128,
        ctx_b, 128, 128LL * 128,
        attn_f, 128, 4096LL * 128,
        4096, 128, 128);

    // 10: merge heads to [b,n,1024] bf16
    merge_attn_kernel<<<dim3(65536), dim3(256), 0, stream>>>(attn_f, attn_b);

    // 11: out = attn @ w_out -> f32 d_out
    gemm_bf16_wmma<<<dim3(8, 128, 1), dim3(256), 0, stream>>>(
        attn_b, 1024, 0,
        wout_b, 1024, 0,
        out, 1024, 0,
        16384, 1024, 1024);
}
